// RefinementHead_14826227106356
// MI455X (gfx1250) — matrix operand order repaired and checked by reading, not compiled, not run on hardware
//
#include <hip/hip_runtime.h>
#include <hip/hip_bf16.h>

typedef __attribute__((ext_vector_type(16))) _Float16 v16h;
typedef __attribute__((ext_vector_type(8)))  float    v8f;

static constexpr int PN    = 1024;  // proposals
static constexpr int NMAXP = 512;   // max points
static constexpr int HIDC  = 256;
static constexpr int FEATC = 512;

// ---------------------------------------------------------------------------
// Pack W1 (9x256 f32) into WMMA B-fragment layout, K zero-padded 9->32.
// Fragment element (nt, lane, e): N = nt*16 + (lane&15), K = 16*(e>>3) + 8*(lane>>4) + (e&7)
// ---------------------------------------------------------------------------
__global__ void pack_w1_kernel(const float* __restrict__ W1, _Float16* __restrict__ B1) {
    int i = blockIdx.x * 256 + threadIdx.x;           // 16*32*16 = 8192
    if (i >= 16 * 32 * 16) return;
    int e    = i & 15;
    int lane = (i >> 4) & 31;
    int nt   = i >> 9;                                // 0..15
    int k = ((e >= 8) ? 16 : 0) + 8 * (lane >> 4) + (e & 7); // 0..31
    int n = nt * 16 + (lane & 15);                    // 0..255
    float v = (k < 9) ? W1[k * HIDC + n] : 0.0f;
    B1[i] = (_Float16)v;
}

// ---------------------------------------------------------------------------
// Pack W2 (256x512 f32) into WMMA B-fragment layout (f16).
// Fragment (kt, nt, lane, e): K = kt*32 + 16*(e>>3) + 8*(lane>>4) + (e&7), N = nt*16 + (lane&15)
// ---------------------------------------------------------------------------
__global__ void pack_w2_kernel(const float* __restrict__ W2, _Float16* __restrict__ B2) {
    int i = blockIdx.x * 256 + threadIdx.x;           // 8*32*32*16 = 131072
    if (i >= 8 * 32 * 32 * 16) return;
    int e    = i & 15;
    int lane = (i >> 4) & 31;
    int nt   = (i >> 9) & 31;
    int kt   = i >> 14;
    int k = kt * 32 + ((e >= 8) ? 16 : 0) + 8 * (lane >> 4) + (e & 7); // 0..255
    int n = nt * 16 + (lane & 15);                    // 0..511
    B2[i] = (_Float16)W2[k * FEATC + n];
}

// ---------------------------------------------------------------------------
// Main fused kernel: one workgroup (8 wave32) per proposal.
// __launch_bounds__(256, 4): 4 waves/SIMD32 -> <=256 VGPRs/wave -> two
// workgroups per WGP so another block's WMMA phase overlaps our VALU phases.
// ---------------------------------------------------------------------------
__global__ __launch_bounds__(256, 4) void refine_head_kernel(
    const float* __restrict__ points,     // P*NMAX*3
    const int*   __restrict__ n_points,   // P
    const float* __restrict__ proposals,  // P*6
    const float* __restrict__ b1,         // 256
    const float* __restrict__ b2,         // 512
    const float* __restrict__ Wc,         // 512
    const float* __restrict__ bc,         // 1
    const float* __restrict__ Wr,         // 512*4
    const float* __restrict__ br,         // 4
    const _Float16* __restrict__ B1,      // packed layer1 B, 8192
    const _Float16* __restrict__ B2,      // packed layer2 B, 131072
    float* __restrict__ out)              // [0,1024): cls, [1024,5120): reg
{
    // A1: X fragments for layer-1 (4 mtiles x 32 lanes x 16 f16)
    __shared__ __align__(32) _Float16 A1[4 * 32 * 16];
    // A2: h1 fragments for layer-2 (4 mtiles x 8 ktiles x 32 lanes x 16 f16)
    __shared__ __align__(32) _Float16 A2[4 * 8 * 32 * 16];
    __shared__ float sb1[HIDC];
    __shared__ float sb2[FEATC];
    __shared__ float featm[FEATC];

    const int p    = blockIdx.x;
    const int t    = threadIdx.x;
    const int lane = t & 31;
    const int wave = t >> 5;
    const int np   = n_points[p];

    if (t < HIDC) sb1[t] = b1[t];
    sb2[t]        = b2[t];
    sb2[t + 256]  = b2[t + 256];

    const float pr0 = proposals[p * 6 + 0], pr1 = proposals[p * 6 + 1];
    const float pr2 = proposals[p * 6 + 2], pr3 = proposals[p * 6 + 3];
    const float pr4 = proposals[p * 6 + 4], pr5 = proposals[p * 6 + 5];

    // Per-wave running column maxima (wave owns feature ntiles wave*4..wave*4+3;
    // lanes 0..15 hold the column maxima after the shfl_xor(16) merge).
    float colmax[4] = {0.f, 0.f, 0.f, 0.f};

    __syncthreads();

    #pragma unroll 1
    for (int cb = 0; cb < NMAXP; cb += 64) {
        if (cb >= np) break;   // np is block-uniform -> uniform branch

        // ---- Phase A0: build X A-fragments (x = [rel3 | proposal6 | 0-pad])
        #pragma unroll
        for (int ii = 0; ii < 8; ++ii) {
            int i  = t * 8 + ii;                 // 0..2047
            int e  = i & 15;
            int ln = (i >> 4) & 31;
            int mt = i >> 9;                     // 0..3
            int M  = ln & 15;
            int k  = ((e >= 8) ? 16 : 0) + 8 * (ln >> 4) + (e & 7);
            int n  = cb + mt * 16 + M;           // point index < 512
            float v;
            if (k < 3) {
                float c = (k == 0) ? pr0 : (k == 1) ? pr1 : pr2;
                v = points[(p * NMAXP + n) * 3 + k] - c;
            } else if (k < 9) {
                v = (k == 3) ? pr0 : (k == 4) ? pr1 : (k == 5) ? pr2
                  : (k == 6) ? pr3 : (k == 7) ? pr4 : pr5;
            } else {
                v = 0.0f;
            }
            A1[i] = (_Float16)v;
        }
        __syncthreads();

        // ---- Phase A1: layer-1 WMMA, bias+ReLU, scatter h1 into A2 fragments
        #pragma unroll 1
        for (int half = 0; half < 2; ++half) {
            const int nt = wave * 2 + half;                 // hidden ntile
            const v16h bf = *(const v16h*)(B1 + (nt * 32 + lane) * 16);
            const float bias = sb1[nt * 16 + (lane & 15)];
            const int h   = nt * 16 + (lane & 15);          // hidden index
            const int kt2 = h >> 5;
            const int kk  = h & 31;
            const int eo  = (kk & 7) + 8 * ((kk >> 4) & 1); // element within frag
            const int lhalf = 16 * ((kk >> 3) & 1);         // lane-half from K
            #pragma unroll 1
            for (int mt = 0; mt < 4; ++mt) {
                v16h a = *(const v16h*)(A1 + (mt * 32 + lane) * 16);
                v8f c = {};
                c = __builtin_amdgcn_wmma_f32_16x16x32_f16(
                        false, a, false, bf, (short)0, c, false, false);
                const int Mbase = 8 * (lane >> 4);
                #pragma unroll
                for (int r = 0; r < 8; ++r) {
                    float v = c[r] + bias;
                    v = v > 0.f ? v : 0.f;
                    int Mrow = r + Mbase;                   // 0..15
                    int dst  = mt * 4096 + kt2 * 512 + (Mrow + lhalf) * 16 + eo;
                    A2[dst] = (_Float16)v;
                }
            }
        }
        __syncthreads();

        // ---- Phase B: layer-2 WMMA (K=256 as 8 steps) + bias+ReLU + max-pool
        #pragma unroll 1
        for (int nti = 0; nti < 4; ++nti) {
            const int nt = wave * 4 + nti;
            v16h Bf[8];
            #pragma unroll
            for (int kt = 0; kt < 8; ++kt)
                Bf[kt] = *(const v16h*)(B2 + ((kt * 32 + nt) * 32 + lane) * 16);
            const float bias = sb2[nt * 16 + (lane & 15)];
            float cm = colmax[nti];
            #pragma unroll 1
            for (int mt = 0; mt < 4; ++mt) {
                v8f c = {};
                #pragma unroll
                for (int kt = 0; kt < 8; ++kt) {
                    v16h a = *(const v16h*)(A2 + mt * 4096 + kt * 512 + lane * 16);
                    c = __builtin_amdgcn_wmma_f32_16x16x32_f16(
                            false, a, false, Bf[kt], (short)0, c, false, false);
                }
                float m = 0.f;
                const int nbase = cb + mt * 16 + 8 * (lane >> 4);
                #pragma unroll
                for (int r = 0; r < 8; ++r) {
                    float v = c[r] + bias;
                    v = v > 0.f ? v : 0.f;
                    if (nbase + r < np) m = fmaxf(m, v);
                }
                m  = fmaxf(m, __shfl_xor(m, 16));  // merge M halves -> lanes 0..15
                cm = fmaxf(cm, m);
            }
            colmax[nti] = cm;
        }
        __syncthreads();
    }

    // ---- Publish pooled features (ReLU>=0 so 0-init matches reference incl. np==0)
    if (lane < 16) {
        #pragma unroll
        for (int nti = 0; nti < 4; ++nti)
            featm[(wave * 4 + nti) * 16 + lane] = colmax[nti];
    }
    __syncthreads();

    // ---- Heads: 5 dot products of length 512, wave 0, deterministic order
    if (t < 32) {
        float o[5];
        #pragma unroll 1
        for (int j = 0; j < 5; ++j) {
            float acc = 0.f;
            for (int i = lane; i < FEATC; i += 32) {
                float w = (j == 0) ? Wc[i] : Wr[i * 4 + (j - 1)];
                acc += featm[i] * w;
            }
            acc += __shfl_xor(acc, 16);
            acc += __shfl_xor(acc, 8);
            acc += __shfl_xor(acc, 4);
            acc += __shfl_xor(acc, 2);
            acc += __shfl_xor(acc, 1);
            o[j] = acc;
        }
        if (lane == 0) {
            out[p] = o[0] + bc[0];
            #pragma unroll
            for (int j = 0; j < 4; ++j)
                out[PN + p * 4 + j] = o[j + 1] + br[j];
        }
    }
}

// ---------------------------------------------------------------------------
extern "C" void kernel_launch(void* const* d_in, const int* in_sizes, int n_in,
                              void* d_out, int out_size, void* d_ws, size_t ws_size,
                              hipStream_t stream) {
    const float* points    = (const float*)d_in[0];
    const int*   n_points  = (const int*)  d_in[1];
    const float* proposals = (const float*)d_in[2];
    const float* W1        = (const float*)d_in[3];
    const float* b1        = (const float*)d_in[4];
    const float* W2        = (const float*)d_in[5];
    const float* b2        = (const float*)d_in[6];
    const float* Wc        = (const float*)d_in[7];
    const float* bc        = (const float*)d_in[8];
    const float* Wr        = (const float*)d_in[9];
    const float* br        = (const float*)d_in[10];

    _Float16* B2 = (_Float16*)d_ws;                               // 131072 f16 = 256 KB
    _Float16* B1 = (_Float16*)((char*)d_ws + 131072 * sizeof(_Float16)); // 8192 f16

    pack_w2_kernel<<<512, 256, 0, stream>>>(W2, B2);
    pack_w1_kernel<<<32, 256, 0, stream>>>(W1, B1);
    refine_head_kernel<<<PN, 256, 0, stream>>>(points, n_points, proposals,
                                               b1, b2, Wc, bc, Wr, br,
                                               B1, B2, (float*)d_out);
}
